// GNNLayer_9448928051210
// MI455X (gfx1250) — matrix operand order repaired and checked by reading, not compiled, not run on hardware
//
#include <hip/hip_runtime.h>

typedef __attribute__((ext_vector_type(16))) _Float16 v16h;
typedef __attribute__((ext_vector_type(8)))  float    v8f;

#define NROWS 262144      // N
#define PATHN 262143      // N-1 path rows
#define DD 512
#define HH 128
#define ROWS_PER_BLOCK 256
#define WSB_HALVES 65536  // 8 n-tiles * 16 ksteps * 32 lanes * 16 halves

__device__ __forceinline__ int imin(int a, int b) { return a < b ? a : b; }

// ---------------------------------------------------------------------------
// k0: zero accumulators, compute tvec = target@aW1[:D]+ab1, and convert
//     aW1[D:] into f16 WMMA B-fragment layout in scratch.
//     wsB layout: frag_id = (t*16 + ks)*32 + lane ; 16 halves per frag,
//     half h of lane holds W[ks*32 + (h&8)*2 + (lane>=16?8:0) + (h&7)][t*16 + (lane&15)].
// ---------------------------------------------------------------------------
__global__ void k0_prep(const float* __restrict__ feat,
                        const float* __restrict__ aW1,
                        const float* __restrict__ ab1,
                        _Float16* __restrict__ wsB,
                        float* __restrict__ wsF)
{
    const int tid = threadIdx.x;
    if (blockIdx.x == 0) {
        // wsF[128] = s accumulator, wsF[129..640] = agg accumulator
        for (int i = tid; i < 513; i += 256) wsF[128 + i] = 0.f;
        if (tid < HH) {
            float acc = ab1[tid];
            for (int k = 0; k < DD; ++k)
                acc = fmaf(feat[k], aW1[k * HH + tid], acc);
            wsF[tid] = acc;   // tvec
        }
    } else {
        const int base = ((blockIdx.x - 1) * 256 + tid) * 4;
        #pragma unroll
        for (int u = 0; u < 4; ++u) {
            const int idx  = base + u;
            const int h    = idx & 15;
            const int lane = (idx >> 4) & 31;
            const int ks   = (idx >> 9) & 15;
            const int t    = idx >> 13;
            const int K    = ks * 32 + ((h & 8) << 1) + ((lane >> 4) << 3) + (h & 7);
            const int n    = t * 16 + (lane & 15);
            wsB[idx] = (_Float16)aW1[(DD + K) * HH + n];
        }
    }
}

// ---------------------------------------------------------------------------
// k1: fused copy + WMMA GEMM + attention + weighted aggregation partials.
//     256 path rows per block; 8 waves, each owns 2 x 16-row tiles.
// ---------------------------------------------------------------------------
__global__ void __launch_bounds__(256)
k1_main(const float* __restrict__ feat,
        const float* __restrict__ aW2,
        const float* __restrict__ ab2,
        const _Float16* __restrict__ wsB,
        float* __restrict__ wsF,
        float* __restrict__ out)
{
    __shared__ float lds_attn[ROWS_PER_BLOCK];
    const int tid  = threadIdx.x;
    const int lane = tid & 31;
    const int wave = tid >> 5;
    const int gr0  = blockIdx.x * ROWS_PER_BLOCK;   // first path row of block

    // -------- Phase 1: coalesced copy of path rows to output (out row = path+1)
    {
        const float4* in4 = (const float4*)feat;
        float4* out4 = (float4*)out;
        const int elems4 = ROWS_PER_BLOCK * (DD / 4);
        for (int i = tid; i < elems4; i += 256) {
            const int p = gr0 + (i >> 7);     // DD/4 == 128
            const int c = i & 127;
            if (p < PATHN) {
                const long o = (long)(p + 1) * (DD / 4) + c;
                out4[o] = in4[o];
            }
        }
    }

    // -------- Phase 2: h_pre = rows @ W via v_wmma_f32_16x16x32_f16
    const int   mhalf = lane & 15;
    const int   koff  = (lane >= 16) ? 8 : 0;
    const float* rp0 = feat + (long)(imin(gr0 + wave * 32 + mhalf,      PATHN - 1) + 1) * DD;
    const float* rp1 = feat + (long)(imin(gr0 + wave * 32 + 16 + mhalf, PATHN - 1) + 1) * DD;

    v8f acc[2][8];
    {
        v8f z = {};
        #pragma unroll
        for (int j = 0; j < 2; ++j)
            #pragma unroll
            for (int t = 0; t < 8; ++t)
                acc[j][t] = z;
    }

    for (int ks = 0; ks < 16; ++ks) {
        const int k0 = ks * 32 + koff;
        v16h a0, a1;
        #pragma unroll
        for (int p = 0; p < 8; ++p) {
            // A layout (16-bit, 16x32): halves 0..7 -> K=k0+0..7, halves 8..15 -> K=k0+16..23
            const int K = k0 + ((p < 4) ? 2 * p : 2 * p + 8);
            const float2 f0 = *(const float2*)(rp0 + K);
            const float2 f1 = *(const float2*)(rp1 + K);
            a0[2 * p]     = (_Float16)f0.x;
            a0[2 * p + 1] = (_Float16)f0.y;
            a1[2 * p]     = (_Float16)f1.x;
            a1[2 * p + 1] = (_Float16)f1.y;
        }
        const v16h* bp = (const v16h*)wsB + (ks * 32 + lane);
        #pragma unroll
        for (int t = 0; t < 8; ++t) {
            const v16h b = bp[t * 512];   // tile stride = 16 ksteps * 32 frags
            acc[0][t] = __builtin_amdgcn_wmma_f32_16x16x32_f16(
                false, a0, false, b, (short)0, acc[0][t], false, false);
            acc[1][t] = __builtin_amdgcn_wmma_f32_16x16x32_f16(
                false, a1, false, b, (short)0, acc[1][t], false, false);
        }
    }

    // -------- Phase 3: attn = sigmoid(relu(h_pre + tvec) . aW2 + ab2)
    float tv[8], w2[8];
    #pragma unroll
    for (int t = 0; t < 8; ++t) {
        tv[t] = wsF[t * 16 + mhalf];      // tvec (incl. ab1)
        w2[t] = aW2[t * 16 + mhalf];
    }
    const float ab2v = ab2[0];
    #pragma unroll
    for (int j = 0; j < 2; ++j) {
        #pragma unroll
        for (int r = 0; r < 8; ++r) {
            float p = 0.f;
            #pragma unroll
            for (int t = 0; t < 8; ++t) {
                float hv = acc[j][t][r] + tv[t];
                hv = fmaxf(hv, 0.f);
                p = fmaf(hv, w2[t], p);
            }
            // reduce across the 16-lane half (C layout: lanes 0-15 = row r, 16-31 = row 8+r)
            p += __shfl_xor(p, 8);
            p += __shfl_xor(p, 4);
            p += __shfl_xor(p, 2);
            p += __shfl_xor(p, 1);
            float at = 1.f / (1.f + __expf(-(p + ab2v)));
            const int lr = wave * 32 + j * 16 + ((lane < 16) ? r : (8 + r));
            if (gr0 + lr >= PATHN) at = 0.f;
            if (mhalf == 0) lds_attn[lr] = at;
        }
    }
    __syncthreads();

    // -------- Phase 4a: block partial of s = sum(attn_unnorm)
    if (tid < 32) {
        float s = 0.f;
        for (int i = tid; i < ROWS_PER_BLOCK; i += 32) s += lds_attn[i];
        s += __shfl_xor(s, 16);
        s += __shfl_xor(s, 8);
        s += __shfl_xor(s, 4);
        s += __shfl_xor(s, 2);
        s += __shfl_xor(s, 1);
        if (tid == 0) atomicAdd(&wsF[128], s);
    }

    // -------- Phase 4b: block partial of agg = sum(attn_unnorm * path_row)
    {
        const int c0 = tid * 2;
        float ax = 0.f, ay = 0.f;
        for (int r = 0; r < ROWS_PER_BLOCK; ++r) {
            const float a = lds_attn[r];                 // 0 for padded rows
            const long row = (long)(imin(gr0 + r, PATHN - 1) + 1);
            const float2 v = *(const float2*)(feat + row * DD + c0);
            ax = fmaf(a, v.x, ax);
            ay = fmaf(a, v.y, ay);
        }
        atomicAdd(&wsF[129 + c0],     ax);
        atomicAdd(&wsF[129 + c0 + 1], ay);
    }
}

// ---------------------------------------------------------------------------
// k2: tiny tail — comb=[target, agg/s], gate/update GEMVs, write out row 0.
// ---------------------------------------------------------------------------
__global__ void __launch_bounds__(512)
k2_final(const float* __restrict__ feat,
         const float* __restrict__ uW1, const float* __restrict__ ub1,
         const float* __restrict__ uW2, const float* __restrict__ ub2,
         const float* __restrict__ gW,  const float* __restrict__ gb,
         const float* __restrict__ wsF, float* __restrict__ out)
{
    __shared__ float comb[2 * DD];
    __shared__ float t1s[DD];
    const int tid = threadIdx.x;
    const float s = wsF[128];
    for (int c = tid; c < 2 * DD; c += 512) {
        float v;
        if (c < DD) v = feat[c];
        else {
            const float a = wsF[129 + c - DD];
            v = (s > 0.f) ? a / s : a;
        }
        comb[c] = v;
    }
    __syncthreads();
    const int n = tid;
    float g = gb[n], u = ub1[n];
    for (int k = 0; k < 2 * DD; ++k) {
        const float cv = comb[k];
        g = fmaf(cv, gW[(long)k * DD + n], g);
        u = fmaf(cv, uW1[(long)k * DD + n], u);
    }
    g = 1.f / (1.f + __expf(-g));
    u = fmaxf(u, 0.f);
    t1s[n] = u;
    __syncthreads();
    float up = ub2[n];
    for (int k = 0; k < DD; ++k)
        up = fmaf(t1s[k], uW2[(long)k * DD + n], up);
    out[n] = feat[n] + g * up;
}

// ---------------------------------------------------------------------------
extern "C" void kernel_launch(void* const* d_in, const int* in_sizes, int n_in,
                              void* d_out, int out_size, void* d_ws, size_t ws_size,
                              hipStream_t stream) {
    const float* feat = (const float*)d_in[0];
    const float* aW1  = (const float*)d_in[1];
    const float* ab1  = (const float*)d_in[2];
    const float* aW2  = (const float*)d_in[3];
    const float* ab2  = (const float*)d_in[4];
    const float* uW1  = (const float*)d_in[5];
    const float* ub1  = (const float*)d_in[6];
    const float* uW2  = (const float*)d_in[7];
    const float* ub2  = (const float*)d_in[8];
    const float* gW   = (const float*)d_in[9];
    const float* gb   = (const float*)d_in[10];
    float* out = (float*)d_out;

    _Float16* wsB = (_Float16*)d_ws;
    float* wsF = (float*)((char*)d_ws + (size_t)WSB_HALVES * 2);  // tvec[128], s, agg[512]

    k0_prep<<<65, 256, 0, stream>>>(feat, aW1, ab1, wsB, wsF);
    const int nblk = (PATHN + ROWS_PER_BLOCK - 1) / ROWS_PER_BLOCK;  // 1024
    k1_main<<<nblk, 256, 0, stream>>>(feat, aW2, ab2, wsB, wsF, out);
    k2_final<<<1, 512, 0, stream>>>(feat, uW1, ub1, uW2, ub2, gW, gb, wsF, out);
}
